// LK_77653008711882
// MI455X (gfx1250) — matrix operand order repaired and checked by reading, not compiled, not run on hardware
//
#include <hip/hip_runtime.h>
#include <hip/hip_bf16.h>
#include <stdint.h>

// ---------------- problem constants ----------------
#define BB   4
#define SS   8
#define CC   3
#define HH   256
#define WW   256
#define PP   68
#define DSC  4
#define HD   64
#define WD   64
#define NIMG (BB*SS)          // 32
#define KTOT 27               // C*3*3
#define MT   5                // ceil(68/16) M tiles
#define NPOS (HD*WD)          // 4096
#define NT   (NPOS/16)        // 256 N tiles per image
#define NSAMP 289             // (2*8+1)^2 LK window
#define LKSTEPS 10

// output offsets (floats, in return order)
#define HEAT_OFF  0
#define HEAT_SZ   (NIMG*PP*NPOS)            // 8,912,896
#define LOCS_OFF  (HEAT_OFF + HEAT_SZ)
#define LOCS_SZ   (NIMG*PP*2)
#define SCOS_OFF  (LOCS_OFF + LOCS_SZ)
#define SCOS_SZ   (NIMG*PP)
#define NEXT_OFF  (SCOS_OFF + SCOS_SZ)
#define FBCK_OFF  (NEXT_OFF + LOCS_SZ)
#define BACK_OFF  (FBCK_OFF + LOCS_SZ)

// workspace byte offsets
#define WS_XH     0                          // fp16 inputs: 6,291,456 halves
#define WS_APACK  12582912                   // 1280 u32 packed A fragments
#define WS_GRAY   12588032                   // fp32 gray: 2,097,152 floats

typedef __attribute__((ext_vector_type(16))) _Float16 v16h;
typedef __attribute__((ext_vector_type(8)))  float    v8f;

#if __has_builtin(__builtin_amdgcn_global_load_async_to_lds_b128)
#define HAVE_ASYNC_LDS 1
// builtin signature (from hipcc diagnostic): pointers to int-vector_size(16)
// in global (AS1) and LDS (AS3) address spaces.
typedef int v4i_t __attribute__((vector_size(16)));
typedef __attribute__((address_space(1))) v4i_t gv4i_t;
typedef __attribute__((address_space(3))) v4i_t lv4i_t;
#else
#define HAVE_ASYNC_LDS 0
#endif

// ---------------- wave32 reductions ----------------
__device__ __forceinline__ float wave_sum32(float v) {
#pragma unroll
  for (int m = 16; m >= 1; m >>= 1) v += __shfl_xor(v, m, 32);
  return v;
}
__device__ __forceinline__ float wave_max32(float v) {
#pragma unroll
  for (int m = 16; m >= 1; m >>= 1) v = fmaxf(v, __shfl_xor(v, m, 32));
  return v;
}

// ---------------- prep kernels ----------------
__global__ __launch_bounds__(256) void cvt_f16_kernel(const float* __restrict__ in,
                                                      _Float16* __restrict__ xh) {
  int i = blockIdx.x * 256 + threadIdx.x;
  xh[i] = (_Float16)in[i];
}

__global__ __launch_bounds__(256) void gray_kernel(const float* __restrict__ in,
                                                   float* __restrict__ gray) {
  int tid = blockIdx.x * 256 + threadIdx.x;     // < 32*65536
  int frame = tid >> 16;
  int pix   = tid & 65535;
  const float* base = in + (size_t)frame * (CC * HH * WW) + pix;
  gray[tid] = 0.299f * base[0] + 0.587f * base[HH*WW] + 0.114f * base[2*HH*WW];
}

// Pack Wdet into the 16-bit 16x32 A-fragment layout:
// lane<16: vgpr i<4 -> K=2i,2i+1 ; i>=4 -> K=16+2(i-4),..
// lane>=16: vgpr i<4 -> K=8+2i ; i>=4 -> K=24+2(i-4)
__global__ __launch_bounds__(256) void packA_kernel(const float* __restrict__ Wdet,
                                                    uint32_t* __restrict__ apack) {
  int idx = blockIdx.x * 256 + threadIdx.x;     // < 1280 = 5*32*8
  int i    = idx & 7;
  int lane = (idx >> 3) & 31;
  int mt   = idx >> 8;
  int m    = lane & 15;
  int p    = mt * 16 + m;
  int kb   = (i < 4) ? ((lane < 16) ? 0 : 8) : ((lane < 16) ? 16 : 24);
  int k0   = kb + 2 * (i & 3);
  float w0 = (p < PP && k0     < KTOT) ? Wdet[p * KTOT + k0]     : 0.0f;
  float w1 = (p < PP && k0 + 1 < KTOT) ? Wdet[p * KTOT + k0 + 1] : 0.0f;
  union { _Float16 h[2]; uint32_t u; } pk;
  pk.h[0] = (_Float16)w0;
  pk.h[1] = (_Float16)w1;
  apack[idx] = pk.u;
}

// ---------------- detector conv via WMMA (implicit GEMM, K=27 -> 32) ----------
// One wave per (image, Ntile); B fragment gathered once, reused across all 5
// M tiles (5 WMMAs per wave) -> im2col gather amortized 5x.
__global__ __launch_bounds__(256) void conv_wmma_kernel(const _Float16* __restrict__ xh,
                                                        const uint32_t* __restrict__ apack,
                                                        const float* __restrict__ bdet,
                                                        float* __restrict__ heat) {
  int wave = (blockIdx.x * 256 + threadIdx.x) >> 5;  // < 32*256
  int lane = threadIdx.x & 31;
  int nt = wave & (NT - 1);
  int n  = wave >> 8;

  // B fragment: im2col column for this lane (lane<16: K 0-7,16-23, lane>=16: K 8-15,24-31)
  int col = lane & 15;
  int pos = nt * 16 + col;
  int xo = pos & (WD - 1);
  int yo = pos >> 6;
  const _Float16* xbase = xh + (size_t)n * (CC * HH * WW);
  int kbase = (lane < 16) ? 0 : 8;
  v16h bfrag;
#pragma unroll
  for (int e = 0; e < 16; ++e) {
    int k = kbase + e + ((e >= 8) ? 8 : 0);
    _Float16 v = (_Float16)0.0f;
    if (k < KTOT) {
      int c = k / 9;
      int r = k - 9 * c;
      int ky = r / 3;
      int kx = r - 3 * ky;
      v = xbase[(c * HH + (DSC * yo + ky)) * WW + (DSC * xo + kx)];
    }
    bfrag[e] = v;
  }

  int mb = (lane < 16) ? 0 : 8;
#pragma unroll
  for (int mt = 0; mt < MT; ++mt) {
    union { uint32_t u[8]; v16h h; } ua;
    const uint32_t* ap = apack + (mt * 32 + lane) * 8;
#pragma unroll
    for (int j = 0; j < 8; ++j) ua.u[j] = ap[j];

    v8f acc = {};
    acc = __builtin_amdgcn_wmma_f32_16x16x32_f16(
        /*neg_a=*/false, ua.h, /*neg_b=*/false, bfrag,
        /*c_mod=*/(short)0, acc, /*reuse_a=*/false, /*reuse_b=*/false);

#pragma unroll
    for (int r2 = 0; r2 < 8; ++r2) {
      int p = mt * 16 + mb + r2;
      if (p < PP) heat[((size_t)n * PP + p) * NPOS + pos] = acc[r2] + bdet[p];
    }
  }
}

// ---------------- softmax + soft-argmax + score ----------------
// Stage the 16KB heatmap row into LDS with CDNA5 async copies (ASYNCcnt),
// then both passes read at LDS bandwidth.
__global__ __launch_bounds__(256) void softargmax_kernel(const float* __restrict__ heat,
                                                         float* __restrict__ locs,
                                                         float* __restrict__ scos) {
  int row = blockIdx.x;                      // n*68+p
  const float* hp = heat + (size_t)row * NPOS;
  int tid = threadIdx.x;
  __shared__ float srow[NPOS];
  __shared__ float sm[8], s0[8], s1[8], s2[8];

#if HAVE_ASYNC_LDS
#pragma unroll
  for (int i = 0; i < 4; ++i) {
    int e = i * 1024 + tid * 4;              // 4 floats (b128) per copy
    __builtin_amdgcn_global_load_async_to_lds_b128(
        (gv4i_t*)(uintptr_t)(hp + e),
        (lv4i_t*)(uint32_t)(uintptr_t)(srow + e),
        0, 0);
  }
#if __has_builtin(__builtin_amdgcn_s_wait_asynccnt)
  __builtin_amdgcn_s_wait_asynccnt(0);
#else
  asm volatile("s_wait_asynccnt 0x0" ::: "memory");
#endif
  __syncthreads();
#else
  for (int i = tid; i < NPOS; i += 256) srow[i] = hp[i];
  __syncthreads();
#endif

  float m = -3.4e38f;
  for (int i = tid; i < NPOS; i += 256) m = fmaxf(m, srow[i]);
  m = wave_max32(m);
  if ((tid & 31) == 0) sm[tid >> 5] = m;
  __syncthreads();
  if (tid == 0) {
    float t = sm[0];
#pragma unroll
    for (int j = 1; j < 8; ++j) t = fmaxf(t, sm[j]);
    sm[0] = t;
  }
  __syncthreads();
  float mx = sm[0];

  float se = 0.f, sx = 0.f, sy = 0.f;
  for (int i = tid; i < NPOS; i += 256) {
    float e = __expf(srow[i] - mx);
    se += e;
    sx += e * (float)(i & (WD - 1));
    sy += e * (float)(i >> 6);
  }
  se = wave_sum32(se); sx = wave_sum32(sx); sy = wave_sum32(sy);
  if ((tid & 31) == 0) { int w = tid >> 5; s0[w] = se; s1[w] = sx; s2[w] = sy; }
  __syncthreads();
  if (tid == 0) {
    float t0 = 0.f, t1 = 0.f, t2 = 0.f;
#pragma unroll
    for (int j = 0; j < 8; ++j) { t0 += s0[j]; t1 += s1[j]; t2 += s2[j]; }
    locs[row * 2 + 0] = t1 / t0 * (float)DSC;
    locs[row * 2 + 1] = t2 / t0 * (float)DSC;
    scos[row] = mx;
  }
}

// ---------------- Lucas-Kanade ----------------
__device__ __forceinline__ float texel(const float* img, int x, int y) {
  return img[y * WW + x];
}
__device__ __forceinline__ float gx_at(const float* img, int x, int y) {
  int xp = (x + 1 < WW) ? x + 1 : WW - 1;
  int xm = (x - 1 > 0) ? x - 1 : 0;
  return 0.5f * (texel(img, xp, y) - texel(img, xm, y));
}
__device__ __forceinline__ float gy_at(const float* img, int x, int y) {
  int yp = (y + 1 < HH) ? y + 1 : HH - 1;
  int ym = (y - 1 > 0) ? y - 1 : 0;
  return 0.5f * (texel(img, x, yp) - texel(img, x, ym));
}

__device__ __forceinline__ void clamp_corners(float x, float y,
                                              int& x0, int& y0, int& x1, int& y1,
                                              float& wx, float& wy) {
  x = fminf(fmaxf(x, 0.0f), (float)WW - 1.001f);
  y = fminf(fmaxf(y, 0.0f), (float)HH - 1.001f);
  float fx0 = floorf(x), fy0 = floorf(y);
  wx = x - fx0; wy = y - fy0;
  x0 = (int)fx0; y0 = (int)fy0;
  x1 = (x0 + 1 < WW) ? x0 + 1 : WW - 1;
  y1 = (y0 + 1 < HH) ? y0 + 1 : HH - 1;
}

__device__ float sample_i(const float* img, float x, float y) {
  int x0, y0, x1, y1; float wx, wy;
  clamp_corners(x, y, x0, y0, x1, y1, wx, wy);
  float v00 = texel(img, x0, y0), v01 = texel(img, x1, y0);
  float v10 = texel(img, x0, y1), v11 = texel(img, x1, y1);
  return (1.f - wy) * ((1.f - wx) * v00 + wx * v01) + wy * ((1.f - wx) * v10 + wx * v11);
}

__device__ void sample_igxgy(const float* img, float x, float y,
                             float& I, float& Gx, float& Gy) {
  int x0, y0, x1, y1; float wx, wy;
  clamp_corners(x, y, x0, y0, x1, y1, wx, wy);
  float a00 = (1.f - wy) * (1.f - wx), a01 = (1.f - wy) * wx;
  float a10 = wy * (1.f - wx),         a11 = wy * wx;
  I  = a00 * texel(img, x0, y0) + a01 * texel(img, x1, y0)
     + a10 * texel(img, x0, y1) + a11 * texel(img, x1, y1);
  Gx = a00 * gx_at(img, x0, y0) + a01 * gx_at(img, x1, y0)
     + a10 * gx_at(img, x0, y1) + a11 * gx_at(img, x1, y1);
  Gy = a00 * gy_at(img, x0, y0) + a01 * gy_at(img, x1, y0)
     + a10 * gy_at(img, x0, y1) + a11 * gy_at(img, x1, y1);
}

// One LK track step for one point; whole wave cooperates (9-10 samples/lane).
__device__ void lk_track(const float* __restrict__ oldf, const float* __restrict__ newf,
                         float& ptx, float& pty) {
  int lane = threadIdx.x & 31;
  // CDNA5: speculative prefetch of the new frame window into GL2
  {
    int cx = min(max((int)ptx, 0), WW - 1);
    int cy = min(max((int)pty, 0), HH - 1);
    __builtin_prefetch(newf + cy * WW + cx, 0, 1);
  }
  float I0a[10], Ixa[10], Iya[10];
  float gxx = 0.f, gxy = 0.f, gyy = 0.f;
#pragma unroll
  for (int j = 0; j < 10; ++j) {
    int s = lane + 32 * j;
    float i0 = 0.f, ix = 0.f, iy = 0.f;
    if (s < NSAMP) {
      float ox = (float)(s % 17) - 8.0f;
      float oy = (float)(s / 17) - 8.0f;
      sample_igxgy(oldf, ptx + ox, pty + oy, i0, ix, iy);
    }
    I0a[j] = i0; Ixa[j] = ix; Iya[j] = iy;
    gxx += ix * ix; gxy += ix * iy; gyy += iy * iy;
  }
  gxx = wave_sum32(gxx); gxy = wave_sum32(gxy); gyy = wave_sum32(gyy);
  float det = gxx * gyy - gxy * gxy + 1e-6f;
  float inv00 = gyy / det, inv01 = -gxy / det, inv11 = gxx / det;

  float vx = 0.f, vy = 0.f;
  for (int it = 0; it < LKSTEPS; ++it) {
    float bx = 0.f, by = 0.f;
#pragma unroll
    for (int j = 0; j < 10; ++j) {
      int s = lane + 32 * j;
      if (s < NSAMP) {
        float ox = (float)(s % 17) - 8.0f;
        float oy = (float)(s / 17) - 8.0f;
        float I1 = sample_i(newf, ptx + ox + vx, pty + oy + vy);
        float err = I0a[j] - I1;
        bx += err * Ixa[j];
        by += err * Iya[j];
      }
    }
    bx = wave_sum32(bx); by = wave_sum32(by);
    vx += inv00 * bx + inv01 * by;
    vy += inv01 * bx + inv11 * by;
  }
  ptx += vx; pty += vy;
}

// Two waves per (b,p): wave A runs forward + forward-backward chains (14 steps,
// the critical path), wave B runs the independent backward chain (7 steps).
__global__ __launch_bounds__(32) void lk_kernel(const float* __restrict__ gray,
                                                const float* __restrict__ locs,
                                                float* __restrict__ nextp,
                                                float* __restrict__ fbackp,
                                                float* __restrict__ backp) {
  int id = blockIdx.x;
  bool bwd = (id >= BB * PP);
  int bp = bwd ? id - BB * PP : id;
  int b = bp / PP;
  int p = bp % PP;
  int lane = threadIdx.x;
  const float* g = gray + (size_t)b * SS * (HH * WW);

  if (!bwd) {
    // forward: start at detector locs of frame 0
    float px = locs[((b * SS + 0) * PP + p) * 2 + 0];
    float py = locs[((b * SS + 0) * PP + p) * 2 + 1];
    if (lane == 0) {
      int o = ((b * SS + 0) * PP + p) * 2;
      nextp[o] = px; nextp[o + 1] = py;
    }
    for (int s = 0; s < SS - 1; ++s) {
      lk_track(g + s * (HH * WW), g + (s + 1) * (HH * WW), px, py);
      if (lane == 0) {
        int o = ((b * SS + s + 1) * PP + p) * 2;
        nextp[o] = px; nextp[o + 1] = py;
      }
    }
    // forward-backward: continue from forward end point back to frame 0
    if (lane == 0) {
      int o = ((b * SS + SS - 1) * PP + p) * 2;
      fbackp[o] = px; fbackp[o + 1] = py;
    }
    for (int i = 0; i < SS - 1; ++i) {
      lk_track(g + (SS - 1 - i) * (HH * WW), g + (SS - 2 - i) * (HH * WW), px, py);
      if (lane == 0) {
        int o = ((b * SS + (SS - 2 - i)) * PP + p) * 2;
        fbackp[o] = px; fbackp[o + 1] = py;
      }
    }
  } else {
    // backward: start at detector locs of last frame (independent chain)
    float px = locs[((b * SS + SS - 1) * PP + p) * 2 + 0];
    float py = locs[((b * SS + SS - 1) * PP + p) * 2 + 1];
    if (lane == 0) {
      int o = ((b * SS + SS - 1) * PP + p) * 2;
      backp[o] = px; backp[o + 1] = py;
    }
    for (int i = 0; i < SS - 1; ++i) {
      lk_track(g + (SS - 1 - i) * (HH * WW), g + (SS - 2 - i) * (HH * WW), px, py);
      if (lane == 0) {
        int o = ((b * SS + (SS - 2 - i)) * PP + p) * 2;
        backp[o] = px; backp[o + 1] = py;
      }
    }
  }
}

// ---------------- launch ----------------
extern "C" void kernel_launch(void* const* d_in, const int* in_sizes, int n_in,
                              void* d_out, int out_size, void* d_ws, size_t ws_size,
                              hipStream_t stream) {
  (void)in_sizes; (void)n_in; (void)out_size; (void)ws_size;
  const float* inputs = (const float*)d_in[0];
  const float* Wdet   = (const float*)d_in[1];
  const float* bdet   = (const float*)d_in[2];
  float* out = (float*)d_out;
  char*  ws  = (char*)d_ws;

  _Float16* xh    = (_Float16*)(ws + WS_XH);
  uint32_t* apack = (uint32_t*)(ws + WS_APACK);
  float*    gray  = (float*)   (ws + WS_GRAY);

  // 1. fp16 copy of inputs (6,291,456 elems)
  cvt_f16_kernel<<<dim3((NIMG*CC*HH*WW)/256), dim3(256), 0, stream>>>(inputs, xh);
  // 2. grayscale (2,097,152 elems)
  gray_kernel<<<dim3((NIMG*HH*WW)/256), dim3(256), 0, stream>>>(inputs, gray);
  // 3. pack weights into WMMA A-fragment layout (1280 dwords)
  packA_kernel<<<dim3(5), dim3(256), 0, stream>>>(Wdet, apack);
  // 4. detector conv: 32 images x 256 Ntiles waves, 5 WMMAs per wave
  conv_wmma_kernel<<<dim3((NIMG*NT)/8), dim3(256), 0, stream>>>(
      xh, apack, bdet, out + HEAT_OFF);
  // 5. softmax + soft-argmax + scores: one block per (n,p) row
  softargmax_kernel<<<dim3(NIMG*PP), dim3(256), 0, stream>>>(
      out + HEAT_OFF, out + LOCS_OFF, out + SCOS_OFF);
  // 6. LK tracking: 2 waves per (b,p): fwd+fback chain, and backward chain
  lk_kernel<<<dim3(2*BB*PP), dim3(32), 0, stream>>>(
      gray, out + LOCS_OFF, out + NEXT_OFF, out + FBCK_OFF, out + BACK_OFF);
}
